// CudaFastWeightSumLinearTransformerLayer_15891378995477
// MI455X (gfx1250) — compile-verified, matrix operand order checked by reading, not run on hardware
//
#include <hip/hip_runtime.h>

#define SLEN   1024
#define BSZ    8
#define DMODEL 512
#define NHEAD  8
#define DHEAD  64
#define ROWS   (SLEN*BSZ)        // 8192
#define QKVN   (3*NHEAD*DHEAD)   // 1536
#define EPS_A  1e-5f
#define LN_EPS 1e-5f

typedef __attribute__((ext_vector_type(2)))  float        v2f;
typedef __attribute__((ext_vector_type(8)))  float        v8f;
typedef __attribute__((ext_vector_type(16))) __bf16       v16bf;
typedef __attribute__((ext_vector_type(8)))  unsigned int v8u;

static __device__ __forceinline__ unsigned short f2bf(float f){
  unsigned u = __float_as_uint(f);
  u += 0x7FFFu + ((u >> 16) & 1u);   // round-to-nearest-even
  return (unsigned short)(u >> 16);
}

static __device__ __forceinline__ v8f wmma_f32(v2f a, v2f b, v8f c){
  return __builtin_amdgcn_wmma_f32_16x16x4_f32(false, a, false, b, (short)0, c, false, false);
}
static __device__ __forceinline__ v8f wmma_bf16(v16bf a, v16bf b, v8f c){
  return __builtin_amdgcn_wmma_f32_16x16x32_bf16(false, a, false, b, (short)0, c, false, false);
}

// CDNA5 async global->LDS copy (ASYNCcnt-tracked, bypasses VGPRs).
// VDST = 32-bit LDS byte offset, VADDR = 64-bit global address, saddr = off.
static __device__ __forceinline__ void async_load_b128(unsigned lds_off, const void* gaddr){
  asm volatile("global_load_async_to_lds_b128 %0, %1, off"
               :: "v"(lds_off), "v"((unsigned long long)(uintptr_t)gaddr)
               : "memory");
}
static __device__ __forceinline__ void wait_async0(){
  asm volatile("s_wait_asynccnt 0x0" ::: "memory");
}

// 16x32 bf16 fragment from LDS (row-major, row stride 32 bf16).
// Element e of lane L: row = L&15, k = (L>>4)*8 + e + (e>=8 ? 8 : 0)
// -> two contiguous 16B chunks at kbase and kbase+16.
static __device__ __forceinline__ v16bf load_frag16(const unsigned short* base, int row, int lane){
  const int kbase = (lane >> 4) * 8;
  const unsigned short* p = base + row*32 + kbase;
  const uint4 lo = *reinterpret_cast<const uint4*>(p);
  const uint4 hi = *reinterpret_cast<const uint4*>(p + 16);
  v8u u = { lo.x, lo.y, lo.z, lo.w, hi.x, hi.y, hi.z, hi.w };
  return __builtin_bit_cast(v16bf, u);
}

// ---------------------------------------------------------------- convert
__global__ void k_f32_to_bf16(const float* __restrict__ x, unsigned short* __restrict__ y, int n){
  int i = (blockIdx.x*blockDim.x + threadIdx.x)*4;
  if (i < n){
    float4 v = *reinterpret_cast<const float4*>(x + i);
    ushort4 o; o.x=f2bf(v.x); o.y=f2bf(v.y); o.z=f2bf(v.z); o.w=f2bf(v.w);
    *reinterpret_cast<ushort4*>(y + i) = o;
  }
}

// ---------------------------------------------------------------- bf16 GEMM
// C[M,N] (f32) = A[M,K] (bf16) @ B[K,N] (bf16).
// Block tile 256x64, 4 waves, wave tile 64x64 (16 WMMA from 4 A-frags x 4 B-frags
// per 32-wide K-step). Double-buffered LDS; A streams in via
// global_load_async_to_lds_b128, B staged transposed for the fragment layout.
__global__ __launch_bounds__(128) void k_gemm_bf16(const unsigned short* __restrict__ A,
                                                   const unsigned short* __restrict__ B,
                                                   float* __restrict__ C,
                                                   int M, int N, int K){
  __shared__ unsigned short As [2][256*32];   // [m][k]     16 KB each
  __shared__ unsigned short BsT[2][64*32];    // [n][k]      4 KB each
  const int tid   = threadIdx.x;
  const int wave  = tid >> 5, lane = tid & 31;
  const int mn    = lane & 15;
  const int mbase = (lane >> 4) * 8;
  const int m0 = blockIdx.x * 256, n0 = blockIdx.y * 64;

  const v8f vz = {0,0,0,0,0,0,0,0};
  v8f acc[4][4] = {{vz,vz,vz,vz},{vz,vz,vz,vz},{vz,vz,vz,vz},{vz,vz,vz,vz}};

  const int kr = tid >> 2, q = tid & 3;       // B staging coords

  // issue async copy of A tile (256 x 32 bf16 = 16KB): 8 x b128 per thread
  auto issueA = [&](int buf, int k0){
    #pragma unroll
    for (int i = 0; i < 8; ++i){
      const int cidx = i*128 + tid;           // 16B chunk id, 0..1023
      const int r = cidx >> 2, cc = cidx & 3;
      const unsigned short* g = A + (size_t)(m0 + r)*K + k0 + cc*8;
      const unsigned lds = (unsigned)(uintptr_t)&As[buf][r*32 + cc*8];
      async_load_b128(lds, g);
    }
  };
  auto loadB = [&](int k0, uint4* p){         // 16 bf16 per thread -> regs
    const unsigned short* src = B + (size_t)(k0 + kr)*N + n0 + q*16;
    p[0] = *reinterpret_cast<const uint4*>(src);
    p[1] = *reinterpret_cast<const uint4*>(src + 8);
  };
  auto storeB = [&](int buf, const uint4* p){ // transposed scatter BsT[n][k]
    unsigned int w[8] = {p[0].x,p[0].y,p[0].z,p[0].w,p[1].x,p[1].y,p[1].z,p[1].w};
    #pragma unroll
    for (int e = 0; e < 16; ++e)
      BsT[buf][(q*16 + e)*32 + kr] = (unsigned short)((w[e>>1] >> ((e&1)*16)) & 0xFFFFu);
  };

  // prologue: stage k-step 0
  uint4 bregs[2];
  issueA(0, 0);
  loadB(0, bregs);
  storeB(0, bregs);
  wait_async0();
  __syncthreads();

  const int nsteps = K >> 5;
  int buf = 0;
  for (int s = 0; s < nsteps; ++s){
    const bool hasnext = (s + 1) < nsteps;
    if (hasnext){                              // overlap DMA + global loads with math
      issueA(buf ^ 1, (s + 1) * 32);
      loadB((s + 1) * 32, bregs);
    }
    // 16 WMMA from 4 A-frags x 4 B-frags
    v16bf af[4];
    #pragma unroll
    for (int sub = 0; sub < 4; ++sub)
      af[sub] = load_frag16(&As[buf][0], wave*64 + sub*16 + mn, lane);
    #pragma unroll
    for (int nt = 0; nt < 4; ++nt){
      v16bf bfr = load_frag16(&BsT[buf][0], nt*16 + mn, lane);
      #pragma unroll
      for (int sub = 0; sub < 4; ++sub)
        acc[sub][nt] = wmma_bf16(af[sub], bfr, acc[sub][nt]);
    }
    if (hasnext){
      storeB(buf ^ 1, bregs);
      wait_async0();                           // async A tile landed in LDS
      __syncthreads();                         // publish both next buffers
      buf ^= 1;
    }
  }

  #pragma unroll
  for (int sub = 0; sub < 4; ++sub)
    #pragma unroll
    for (int nt = 0; nt < 4; ++nt)
      #pragma unroll
      for (int r = 0; r < 8; ++r)
        C[(size_t)(m0 + wave*64 + sub*16 + mbase + r)*N + n0 + nt*16 + mn] = acc[sub][nt][r];
}

// ---------------------------------------------------------------- causal linear attention
// One block per (b,h); 4 waves. Chunked form, chunk = 16 = WMMA M tile.
// State S[64][64] f32 in LDS; all matmuls via v_wmma_f32_16x16x4_f32.
__global__ __launch_bounds__(128) void k_linattn(const float* __restrict__ qkv,
                                                 unsigned short* __restrict__ out_bf){
  __shared__ float S[64][64];
  __shared__ float Qf[16][64], Kf[16][64], Vf[16][64];
  __shared__ float Am[16][16];
  __shared__ float zvec[64];
  __shared__ float denom[16];
  __shared__ float red_q[16][8], red_k[16][8], red_di[16][8];

  const int tid   = threadIdx.x;
  const int wave  = tid >> 5, lane = tid & 31;
  const int mn    = lane & 15;
  const int g2    = lane >> 4;
  const int mbase = g2 * 8;
  const int bb = blockIdx.x >> 3;
  const int hh = blockIdx.x & 7;
  const float scale = 0.125f;           // 1/sqrt(64)
  const v8f vz = {0,0,0,0,0,0,0,0};

  for (int i = tid; i < 64*64; i += 128) (&S[0][0])[i] = 0.f;
  if (tid < 64) zvec[tid] = 0.f;
  __syncthreads();

  const int row16 = tid >> 3;   // 0..15
  const int part  = tid & 7;    // 0..7
  const int j0    = part * 8;

  for (int c = 0; c < SLEN/16; ++c){
    // ---- phase A: elu+1 feature map, sum-normalize, inter-chunk denom dot ----
    const int t0 = c*16 + row16;
    const float* src = qkv + (size_t)(t0*BSZ + bb)*QKVN + hh*192;
    float q[8], k[8], v[8];
    float sq = 0.f, sk = 0.f;
    #pragma unroll
    for (int i = 0; i < 8; ++i){
      float x = src[j0 + i];
      float y = src[64 + j0 + i];
      v[i] = src[128 + j0 + i];
      q[i] = (x > 0.f) ? x + 1.f : __expf(x);   // elu(x)+1
      k[i] = (y > 0.f) ? y + 1.f : __expf(y);
      sq += q[i]; sk += k[i];
    }
    red_q[row16][part] = sq;
    red_k[row16][part] = sk;
    __syncthreads();                                        // b1
    float tq = 0.f, tk = 0.f;
    #pragma unroll
    for (int p = 0; p < 8; ++p){ tq += red_q[row16][p]; tk += red_k[row16][p]; }
    const float rq = 1.f / tq, rk = 1.f / tk;
    float di = 0.f;
    #pragma unroll
    for (int i = 0; i < 8; ++i){
      float qn = q[i]*rq, kn = k[i]*rk;
      Qf[row16][j0+i] = qn;
      Kf[row16][j0+i] = kn;
      Vf[row16][j0+i] = v[i];
      di += qn * zvec[j0+i];                                // q_t . z_prev
    }
    red_di[row16][part] = di;
    __syncthreads();                                        // b2

    // ---- scores A = Q K^T (16x16, K=64), computed by every wave ----
    v8f sc = vz;
    #pragma unroll
    for (int kk = 0; kk < 16; ++kk){
      const int kx = kk*4 + g2*2;
      v2f a = { Qf[mn][kx], Qf[mn][kx+1] };                 // A[m][k], m=mn
      v2f b = { Kf[mn][kx], Kf[mn][kx+1] };                 // B[k][n]=Kf[n][k], n=mn
      sc = wmma_f32(a, b, sc);
    }
    #pragma unroll
    for (int r = 0; r < 8; ++r)
      if (mn > mbase + r) sc[r] = 0.f;                      // causal: keep n<=m
    if (wave == 0){
      #pragma unroll
      for (int r = 0; r < 8; ++r) Am[mbase + r][mn] = sc[r];
      float rs[8];
      #pragma unroll
      for (int r = 0; r < 8; ++r) rs[r] = sc[r];
      #pragma unroll
      for (int off = 1; off < 16; off <<= 1){
        #pragma unroll
        for (int r = 0; r < 8; ++r) rs[r] += __shfl_xor(rs[r], off);
      }
      if (mn == 0){                                         // lanes 0 and 16
        #pragma unroll
        for (int r = 0; r < 8; ++r){
          const int m = mbase + r; float d = rs[r];
          #pragma unroll
          for (int p = 0; p < 8; ++p) d += red_di[m][p];
          denom[m] = d;
        }
      }
    }
    if (tid < 64){                                          // z += colsum(K_chunk)
      float a = zvec[tid];
      #pragma unroll
      for (int r = 0; r < 16; ++r) a += Kf[r][tid];
      zvec[tid] = a;
    }
    __syncthreads();                                        // b3

    // ---- O = Q @ S  +  Am @ V   (each wave owns a 16-col quarter) ----
    const int nabs = wave*16 + mn;
    v8f o = vz;
    #pragma unroll
    for (int kk = 0; kk < 16; ++kk){
      const int kx = kk*4 + g2*2;
      v2f a = { Qf[mn][kx], Qf[mn][kx+1] };
      v2f b = { S[kx][nabs], S[kx+1][nabs] };
      o = wmma_f32(a, b, o);
    }
    #pragma unroll
    for (int kk = 0; kk < 4; ++kk){
      const int kx = kk*4 + g2*2;
      v2f a = { Am[mn][kx], Am[mn][kx+1] };
      v2f b = { Vf[kx][nabs], Vf[kx+1][nabs] };
      o = wmma_f32(a, b, o);
    }
    #pragma unroll
    for (int r = 0; r < 8; ++r){
      const int m = mbase + r;
      const int t = c*16 + m;
      const float val = o[r] * scale / (denom[m] + EPS_A);
      out_bf[(size_t)(t*BSZ + bb)*DMODEL + hh*64 + nabs] = f2bf(val);
    }
    __syncthreads();                                        // b4 (S reads done)

    // ---- S += K^T @ V   (wave updates its 16-row strip, all 64 cols) ----
    #pragma unroll
    for (int nt = 0; nt < 4; ++nt){
      v8f ds = vz;
      #pragma unroll
      for (int kk = 0; kk < 4; ++kk){
        const int kx = kk*4 + g2*2;
        v2f a = { Kf[kx][wave*16 + mn], Kf[kx+1][wave*16 + mn] }; // (K^T)[m][k]
        v2f b = { Vf[kx][nt*16 + mn],   Vf[kx+1][nt*16 + mn] };
        ds = wmma_f32(a, b, ds);
      }
      #pragma unroll
      for (int r = 0; r < 8; ++r)
        S[wave*16 + mbase + r][nt*16 + mn] += ds[r];
    }
    __syncthreads();                                        // b5 (chunk end)
  }
}

// ---------------------------------------------------------------- residual + LayerNorm
__global__ __launch_bounds__(256) void k_add_ln(const float* __restrict__ h,
                                                const float* __restrict__ attn,
                                                const float* __restrict__ gamma,
                                                const float* __restrict__ beta,
                                                float* __restrict__ out){
  __shared__ float red[256];
  const int row = blockIdx.x;
  const int tid = threadIdx.x;
  const size_t base = (size_t)row * DMODEL;
  const float x0 = h[base + tid]       + attn[base + tid];
  const float x1 = h[base + tid + 256] + attn[base + tid + 256];
  red[tid] = x0 + x1;
  __syncthreads();
  for (int s = 128; s > 0; s >>= 1){
    if (tid < s) red[tid] += red[tid + s];
    __syncthreads();
  }
  const float mu = red[0] * (1.f / DMODEL);
  __syncthreads();
  const float d0 = x0 - mu, d1 = x1 - mu;
  red[tid] = d0*d0 + d1*d1;
  __syncthreads();
  for (int s = 128; s > 0; s >>= 1){
    if (tid < s) red[tid] += red[tid + s];
    __syncthreads();
  }
  const float rstd = rsqrtf(red[0] * (1.f / DMODEL) + LN_EPS);
  out[base + tid]       = d0 * rstd * gamma[tid]       + beta[tid];
  out[base + tid + 256] = d1 * rstd * gamma[tid + 256] + beta[tid + 256];
}

// ---------------------------------------------------------------- launch
extern "C" void kernel_launch(void* const* d_in, const int* in_sizes, int n_in,
                              void* d_out, int out_size, void* d_ws, size_t ws_size,
                              hipStream_t stream){
  const float* h     = (const float*)d_in[0];
  const float* W_qkv = (const float*)d_in[1];
  const float* W_o   = (const float*)d_in[2];
  const float* gamma = (const float*)d_in[3];
  const float* beta  = (const float*)d_in[4];

  char* ws = (char*)d_ws;
  size_t off = 0;
  auto alloc = [&](size_t bytes) -> void* {
    void* p = ws + off;
    off = (off + bytes + 255) & ~size_t(255);
    return p;
  };
  unsigned short* h_bf    = (unsigned short*)alloc((size_t)ROWS*DMODEL*2);   // 8 MB
  unsigned short* wqkv_bf = (unsigned short*)alloc((size_t)DMODEL*QKVN*2);   // 1.5 MB
  unsigned short* wo_bf   = (unsigned short*)alloc((size_t)DMODEL*DMODEL*2); // 0.5 MB
  float*          qkv     = (float*)alloc((size_t)ROWS*QKVN*sizeof(float));  // 48 MB
  unsigned short* lo_bf   = (unsigned short*)alloc((size_t)ROWS*DMODEL*2);   // 8 MB
  float*          attn    = (float*)alloc((size_t)ROWS*DMODEL*sizeof(float));// 16 MB

  k_f32_to_bf16<<<(ROWS*DMODEL/4 + 255)/256, 256, 0, stream>>>(h, h_bf, ROWS*DMODEL);
  k_f32_to_bf16<<<(DMODEL*QKVN/4 + 255)/256, 256, 0, stream>>>(W_qkv, wqkv_bf, DMODEL*QKVN);
  k_f32_to_bf16<<<(DMODEL*DMODEL/4 + 255)/256, 256, 0, stream>>>(W_o, wo_bf, DMODEL*DMODEL);

  k_gemm_bf16<<<dim3(ROWS/256, QKVN/64), 128, 0, stream>>>(h_bf, wqkv_bf, qkv, ROWS, QKVN, DMODEL);

  k_linattn<<<BSZ*NHEAD, 128, 0, stream>>>(qkv, lo_bf);

  k_gemm_bf16<<<dim3(ROWS/256, DMODEL/64), 128, 0, stream>>>(lo_bf, wo_bf, attn, ROWS, DMODEL, DMODEL);

  k_add_ln<<<ROWS, 256, 0, stream>>>(h, attn, gamma, beta, (float*)d_out);
}
